// Emu3VisionVQVectorQuantizer_12137577579080
// MI455X (gfx1250) — matrix operand order repaired and checked by reading, not compiled, not run on hardware
//
#include <hip/hip_runtime.h>
#include <hip/hip_bf16.h>

// Problem constants (match reference: B=2, T=1, C=256, H=64, W=64, K=16384)
#define NROWS   8192      // B*T*H*W
#define CDIM    256
#define KCODES  16384
#define HW      4096      // H*W (spatial stride between channels in x)
#define NTILES  (KCODES / 16)   // 1024 code tiles of 16
#define KSTEPS  (CDIM / 32)     // 8 WMMA k-steps of 32 channels

typedef __attribute__((ext_vector_type(16))) __bf16 v16bf;
typedef __attribute__((ext_vector_type(8)))  float  v8f;

// ---------------------------------------------------------------------------
// Prep 1: per-code squared norms e_sq[k] = sum_c e[k][c]^2
// ---------------------------------------------------------------------------
__global__ __launch_bounds__(256) void esq_kernel(const float* __restrict__ cb,
                                                  float* __restrict__ esq) {
    int k = blockIdx.x * blockDim.x + threadIdx.x;
    if (k >= KCODES) return;
    const float* e = cb + (size_t)k * CDIM;
    float s = 0.f;
    #pragma unroll 8
    for (int c = 0; c < CDIM; ++c) { float v = e[c]; s = fmaf(v, v, s); }
    esq[k] = s;
}

// ---------------------------------------------------------------------------
// Prep 2: swizzle codebook into per-lane WMMA B-fragment order, bf16 hi + lo.
// Flat dest element u encodes: j = u&15 (elem in lane), L = (u>>4)&31 (lane),
// s = (u>>9)&7 (k-step), t = u>>12 (code tile).
// B order used here (contiguous k, same as A): lane L holds column n = L&15,
// element j holds channel k = 16*(L>>4) + j within the 32-channel step.
// ---------------------------------------------------------------------------
__global__ __launch_bounds__(256) void eswz_kernel(const float* __restrict__ cb,
                                                   __bf16* __restrict__ ehi,
                                                   __bf16* __restrict__ elo) {
    unsigned u = blockIdx.x * 256u + threadIdx.x;       // < KCODES*CDIM = 4194304
    int j = u & 15;
    int L = (u >> 4) & 31;
    int s = (u >> 9) & 7;
    int t = u >> 12;
    int code = t * 16 + (L & 15);
    int chan = s * 32 + ((L >> 4) << 4) + j;
    float v = cb[(size_t)code * CDIM + chan];
    __bf16 hi = (__bf16)v;
    ehi[u] = hi;
    elo[u] = (__bf16)(v - (float)hi);
}

// ---------------------------------------------------------------------------
// Main kernel: one workgroup (8 waves) per 16-row M-tile. Each wave keeps all
// A fragments (hi/lo for 8 k-steps) resident in VGPRs and streams code tiles,
// doing 3 bf16 WMMAs per k-step (hi*hi + hi*lo + lo*hi -> ~fp32 accuracy).
// C/D layout: lane holds column n = L&15; VGPR r holds row m = r + 8*(L>>4).
// ---------------------------------------------------------------------------
__global__ __launch_bounds__(256) void vq_wmma_kernel(const float* __restrict__ x,
                                                      const __bf16* __restrict__ ehi,
                                                      const __bf16* __restrict__ elo,
                                                      const float* __restrict__ esq,
                                                      float* __restrict__ out) {
    const int tid  = threadIdx.x;
    const int wave = tid >> 5;
    const int lane = tid & 31;
    const int h    = lane >> 4;
    const int l16  = lane & 15;
    const int wg   = blockIdx.x;            // 0..511 M-tiles
    const int row  = wg * 16 + l16;         // x row owned by this lane in A
    const int nmaj = row >> 12;             // row / HW
    const int nmin = row & (HW - 1);        // row % HW

    // Build resident A fragments.  A order used here (contiguous k, same as B):
    // lane L holds row m = L&15; element j holds channel
    // k = 16*(L>>4) + j within the 32-channel step.
    v16bf ahi[KSTEPS], alo[KSTEPS];
    #pragma unroll
    for (int s = 0; s < KSTEPS; ++s) {
        #pragma unroll
        for (int j = 0; j < 16; ++j) {
            int chan = s * 32 + 16 * h + j;
            float v = x[((size_t)(nmaj * CDIM + chan)) * HW + nmin];
            __bf16 hi2 = (__bf16)v;
            ahi[s][j] = hi2;
            alo[s][j] = (__bf16)(v - (float)hi2);
        }
    }

    const v16bf* Bh = (const v16bf*)ehi;    // fragment-granular (32B) view
    const v16bf* Bl = (const v16bf*)elo;

    float bestD[8];
    int   bestI[8];
    #pragma unroll
    for (int r = 0; r < 8; ++r) { bestD[r] = 3.4e38f; bestI[r] = 0; }

    for (int t = wave; t < NTILES; t += 8) {
        v8f acc = {};
        #pragma unroll
        for (int s = 0; s < KSTEPS; ++s) {
            v16bf bh = Bh[(size_t)t * 256 + s * 32 + lane];
            v16bf bl = Bl[(size_t)t * 256 + s * 32 + lane];
            acc = __builtin_amdgcn_wmma_f32_16x16x32_bf16(
                      false, ahi[s], false, bh, (short)0, acc, false, false);
            acc = __builtin_amdgcn_wmma_f32_16x16x32_bf16(
                      false, ahi[s], false, bl, (short)0, acc, false, false);
            acc = __builtin_amdgcn_wmma_f32_16x16x32_bf16(
                      false, alo[s], false, bh, (short)0, acc, false, false);
        }
        int   code = t * 16 + l16;
        float es   = esq[code];
        #pragma unroll
        for (int r = 0; r < 8; ++r) {
            float d = es - 2.0f * acc[r];
            if (d < bestD[r]) { bestD[r] = d; bestI[r] = code; }
        }
    }

    // Cross-lane argmin within each 16-lane half (columns of the tile).
    #pragma unroll
    for (int r = 0; r < 8; ++r) {
        float d = bestD[r];
        int   i = bestI[r];
        #pragma unroll
        for (int m = 1; m < 16; m <<= 1) {
            float od = __shfl_xor(d, m, 32);
            int   oi = __shfl_xor(i, m, 32);
            if (od < d || (od == d && oi < i)) { d = od; i = oi; }
        }
        bestD[r] = d; bestI[r] = i;
    }

    // Cross-wave combine in LDS: wave w wrote rows r+8*h from lanes 0 and 16.
    __shared__ float sD[8][16];
    __shared__ int   sI[8][16];
    if (l16 == 0) {
        #pragma unroll
        for (int r = 0; r < 8; ++r) {
            sD[wave][r + 8 * h] = bestD[r];
            sI[wave][r + 8 * h] = bestI[r];
        }
    }
    __syncthreads();
    if (tid < 16) {
        float d = sD[0][tid];
        int   i = sI[0][tid];
        #pragma unroll
        for (int w = 1; w < 8; ++w) {
            float od = sD[w][tid];
            int   oi = sI[w][tid];
            if (od < d || (od == d && oi < i)) { d = od; i = oi; }
        }
        out[wg * 16 + tid] = (float)i;   // output row order == (b,t,h,w) flat
    }
}

// ---------------------------------------------------------------------------
// Workspace-free exact fallback (one block per x-row), used only if ws_size
// is too small for the swizzled codebook.
// ---------------------------------------------------------------------------
__global__ __launch_bounds__(256) void vq_fallback_kernel(const float* __restrict__ x,
                                                          const float* __restrict__ cb,
                                                          float* __restrict__ out) {
    __shared__ float xr[CDIM];
    __shared__ float rD[256];
    __shared__ int   rI[256];
    int row  = blockIdx.x;
    int nmaj = row >> 12, nmin = row & (HW - 1);
    for (int c = threadIdx.x; c < CDIM; c += 256)
        xr[c] = x[((size_t)(nmaj * CDIM + c)) * HW + nmin];
    __syncthreads();
    float best = 3.4e38f; int bi = 0;
    for (int k = threadIdx.x; k < KCODES; k += 256) {
        const float* e = cb + (size_t)k * CDIM;
        float es = 0.f, dot = 0.f;
        #pragma unroll 8
        for (int c = 0; c < CDIM; ++c) {
            float ev = e[c];
            es  = fmaf(ev, ev, es);
            dot = fmaf(xr[c], ev, dot);
        }
        float d = es - 2.f * dot;
        if (d < best || (d == best && k < bi)) { best = d; bi = k; }
    }
    rD[threadIdx.x] = best; rI[threadIdx.x] = bi;
    __syncthreads();
    for (int s = 128; s > 0; s >>= 1) {
        if (threadIdx.x < (unsigned)s) {
            float od = rD[threadIdx.x + s];
            int   oi = rI[threadIdx.x + s];
            if (od < rD[threadIdx.x] ||
                (od == rD[threadIdx.x] && oi < rI[threadIdx.x])) {
                rD[threadIdx.x] = od; rI[threadIdx.x] = oi;
            }
        }
        __syncthreads();
    }
    if (threadIdx.x == 0) out[row] = (float)rI[0];
}

// ---------------------------------------------------------------------------
extern "C" void kernel_launch(void* const* d_in, const int* in_sizes, int n_in,
                              void* d_out, int out_size, void* d_ws, size_t ws_size,
                              hipStream_t stream) {
    const float* x  = (const float*)d_in[0];   // (2,1,256,64,64) f32
    const float* cb = (const float*)d_in[1];   // (16384,256) f32
    float* out = (float*)d_out;                // 8192 indices (as float)

    const size_t ehi_bytes = (size_t)KCODES * CDIM * sizeof(__bf16);  // 8 MB
    const size_t need = 2 * ehi_bytes + (size_t)KCODES * sizeof(float);

    if (ws_size >= need) {
        __bf16* ehi = (__bf16*)d_ws;
        __bf16* elo = (__bf16*)((char*)d_ws + ehi_bytes);
        float*  esq = (float*)((char*)d_ws + 2 * ehi_bytes);

        esq_kernel<<<KCODES / 256, 256, 0, stream>>>(cb, esq);
        eswz_kernel<<<(KCODES * CDIM) / 256, 256, 0, stream>>>(cb, ehi, elo);
        vq_wmma_kernel<<<NROWS / 16, 256, 0, stream>>>(x, ehi, elo, esq, out);
    } else {
        vq_fallback_kernel<<<NROWS, 256, 0, stream>>>(x, cb, out);
    }
}
